// MixtureOfExpertsFeedForward_6734508720763
// MI455X (gfx1250) — matrix operand order repaired and checked by reading, not compiled, not run on hardware
//
#include <hip/hip_runtime.h>

// Problem constants (B=4, S=1024 -> N=4096 tokens)
#define NTOK 4096
#define DDIM 1024
#define FDIM 4096
#define NEXP 8
#define NSLOT (NTOK * 2)   // top-2 slots per token

#define SA_STRIDE 40       // halves per LDS row (8-half pad vs 32)
#define TILE_HALVES (128 * SA_STRIDE)

#if defined(__HIP_DEVICE_COMPILE__) &&                                  \
    __has_builtin(__builtin_amdgcn_global_load_async_to_lds_b128) &&    \
    __has_builtin(__builtin_amdgcn_s_wait_asynccnt)
#define USE_ASYNC_LDS 1
#else
#define USE_ASYNC_LDS 0
#endif

typedef __attribute__((ext_vector_type(16))) __bf16 v16bf;
typedef __attribute__((ext_vector_type(8)))  __bf16 v8bf;
typedef __attribute__((ext_vector_type(4)))  __bf16 v4bf;
typedef __attribute__((ext_vector_type(8)))  float  v8f;
typedef __attribute__((ext_vector_type(4)))  int    v4i;
#if USE_ASYNC_LDS
typedef __attribute__((address_space(1))) v4i g_v4i;  // global int4
typedef __attribute__((address_space(3))) v4i l_v4i;  // LDS int4
#endif

__device__ __forceinline__ unsigned f2bfbits(float f) {
  union { float f; unsigned u; } c; c.f = f;
  return (c.u + 0x7FFFu + ((c.u >> 16) & 1u)) >> 16;   // RNE, 16-bit result
}
__device__ __forceinline__ __bf16 f2bf(float f) {
  unsigned short h = (unsigned short)f2bfbits(f);
  return __builtin_bit_cast(__bf16, h);
}
__device__ __forceinline__ float bf2f(__bf16 b) {
  unsigned short h = __builtin_bit_cast(unsigned short, b);
  union { unsigned u; float f; } c; c.u = ((unsigned)h) << 16;
  return c.f;
}

// ---------------- small helper kernels ----------------

__global__ void k_zero(int* counts) {
  if (threadIdx.x < NEXP) counts[threadIdx.x] = 0;
}

__global__ void k_cvt_x(const float* __restrict__ x, __bf16* __restrict__ xb) {
  long i = ((long)blockIdx.x * 256 + threadIdx.x) * 4;
  float4 v = *(const float4*)(x + i);
  v4bf o;
  o[0] = f2bf(v.x); o[1] = f2bf(v.y); o[2] = f2bf(v.z); o[3] = f2bf(v.w);
  *(v4bf*)(xb + i) = o;
}

// One wave32 per token: logits -> softmax -> top-2 (ties keep lowest index,
// matching jax top_k), plus per-expert counts.
__global__ void k_router(const float* __restrict__ x, const float* __restrict__ Wr,
                         const float* __restrict__ br,
                         float* __restrict__ topk_p, int* __restrict__ topk_e,
                         int* __restrict__ counts) {
  const int wv   = threadIdx.x >> 5;
  const int lane = threadIdx.x & 31;
  const int n    = blockIdx.x * 8 + wv;

  float acc[NEXP];
#pragma unroll
  for (int e = 0; e < NEXP; ++e) acc[e] = 0.f;

  const float* xr = x + (long)n * DDIM;
  for (int d = lane; d < DDIM; d += 32) {
    float xv = xr[d];
#pragma unroll
    for (int e = 0; e < NEXP; ++e) acc[e] += xv * Wr[d * NEXP + e];
  }
#pragma unroll
  for (int e = 0; e < NEXP; ++e) {
#pragma unroll
    for (int off = 16; off > 0; off >>= 1)
      acc[e] += __shfl_xor(acc[e], off, 32);
  }
  if (lane == 0) {
    float mx = -1e30f;
#pragma unroll
    for (int e = 0; e < NEXP; ++e) { acc[e] += br[e]; mx = fmaxf(mx, acc[e]); }
    float s = 0.f;
#pragma unroll
    for (int e = 0; e < NEXP; ++e) { acc[e] = __expf(acc[e] - mx); s += acc[e]; }
    float inv = 1.f / s;
    int b1 = 0; float p1 = -1.f;
#pragma unroll
    for (int e = 0; e < NEXP; ++e) { float p = acc[e] * inv; if (p > p1) { p1 = p; b1 = e; } }
    int b2 = 0; float p2 = -1.f;
#pragma unroll
    for (int e = 0; e < NEXP; ++e) {
      if (e == b1) continue;
      float p = acc[e] * inv; if (p > p2) { p2 = p; b2 = e; }
    }
    topk_p[n * 2 + 0] = p1; topk_p[n * 2 + 1] = p2;
    topk_e[n * 2 + 0] = b1; topk_e[n * 2 + 1] = b2;
    atomicAdd(&counts[b1], 1);
    atomicAdd(&counts[b2], 1);
  }
}

__global__ void k_scan(const int* __restrict__ counts, int* __restrict__ offs,
                       int* __restrict__ cursor) {
  if (threadIdx.x == 0) {
    int s = 0;
    for (int e = 0; e < NEXP; ++e) { offs[e] = s; cursor[e] = s; s += counts[e]; }
  }
}

__global__ void k_scatter(const int* __restrict__ topk_e, int* __restrict__ cursor,
                          int* __restrict__ perm) {
  int s = blockIdx.x * 256 + threadIdx.x;
  if (s < NSLOT) {
    int e = topk_e[s];
    int pos = atomicAdd(&cursor[e], 1);
    perm[pos] = s;  // group-position order is nondeterministic; per-slot values are not
  }
}

// hcomb[n] = bf16( habuf[2n] + habuf[2n+1] )
__global__ void k_hcomb(const __bf16* __restrict__ hab, __bf16* __restrict__ hc) {
  const int n = blockIdx.x;
  const __bf16* a = hab + (long)(2 * n) * FDIM;
  const __bf16* b = hab + (long)(2 * n + 1) * FDIM;
  __bf16* o = hc + (long)n * FDIM;
  for (int f = threadIdx.x * 8; f < FDIM; f += 256 * 8) {
    v8bf va = *(const v8bf*)(a + f);
    v8bf vb = *(const v8bf*)(b + f);
    v8bf vo;
#pragma unroll
    for (int j = 0; j < 8; ++j) vo[j] = f2bf(bf2f(va[j]) + bf2f(vb[j]));
    *(v8bf*)(o + f) = vo;
  }
}

// y[n] = yb[2n] + yb[2n+1]
__global__ void k_ycomb(const float* __restrict__ yb, float* __restrict__ y) {
  const int n = blockIdx.x;
  const float* a = yb + (long)(2 * n) * DDIM;
  const float* b = yb + (long)(2 * n + 1) * DDIM;
  float* o = y + (long)n * DDIM;
  for (int d = threadIdx.x; d < DDIM; d += 256) o[d] = a[d] + b[d];
}

// ---------------- grouped expert GEMM (bf16 WMMA, f32 accum) ----------------
// 128x128 tile / block (8 waves; each wave 64x32 = 4x2 WMMA tiles), K-step 32,
// double-buffered LDS, ONE barrier per K-step. A tile goes global->LDS via the
// CDNA5 async DMA path (ASYNCcnt) when available; B tile is fp32->bf16
// converted + transposed through registers (DScnt, covered by __syncthreads).
// STAGE_IN : Hout[slot][f] = bf16(relu(xb[tok] @ W_in[e] + b_in[e]))
// !STAGE_IN: Yout[slot][d] = p_slot * (hcomb[tok] @ W_out[e] + b_out[e])

template <int KD, int ND, bool STAGE_IN>
__global__ __launch_bounds__(256)
void moe_gemm(const __bf16* __restrict__ Amat, const float* __restrict__ Wmat,
              const float* __restrict__ bias, const int* __restrict__ perm,
              const int* __restrict__ offs, const int* __restrict__ counts,
              const float* __restrict__ topk_p, __bf16* __restrict__ Hout,
              float* __restrict__ Yout) {
  __shared__ __bf16 sA[2][TILE_HALVES];  // [row][k], padded rows
  __shared__ __bf16 sB[2][TILE_HALVES];  // [ncol][k] (transposed weight tile)

  const int e    = blockIdx.z;
  const int gcnt = counts[e];
  const int m0   = blockIdx.y * 128;
  if (m0 >= gcnt) return;  // uniform early-exit keeps EXEC all-1s for WMMA
  const int gbase = offs[e];
  const int n0    = blockIdx.x * 128;

  const int tid  = threadIdx.x;
  const int lane = tid & 31;
  const int wv   = tid >> 5;
  const int wm   = wv >> 2;  // 0..1 : 64-row band
  const int wn   = wv & 3;   // 0..3 : 32-col band

  // A staging: 2 threads per row, 16 halves (32B) each
  const int arow = tid >> 1;
  const int acol = (tid & 1) * 16;
  int atok = 0;
  {
    int gm = m0 + arow;
    if (gm < gcnt) atok = perm[gbase + gm] >> 1;  // slot -> token
  }
  const __bf16* aptr = Amat + (long)atok * KD;

  // B staging: thread owns k-pair (2 rows) x 8 consecutive f -> 8 packed b32 stores
  const int bkp = tid >> 4;          // 0..15 -> k = 2*bkp
  const int bk2 = bkp * 2;
  const int bfc = (tid & 15) * 8;    // f chunk
  const float* WepBase = Wmat + (long)e * KD * ND + n0 + bfc;

  // register prefetch (B always; A only on the non-async path)
  uint4 ra0, ra1;
  float4 rb0, rb1, rb2, rb3;

  auto fetchA = [&](int kk) {
    const uint4* g = (const uint4*)(aptr + kk + acol);
    ra0 = g[0];
    ra1 = g[1];
  };
  auto stageA = [&](int buf) {
    __bf16* sa = &sA[buf][0];
    *(uint4*)(&sa[arow * SA_STRIDE + acol])     = ra0;
    *(uint4*)(&sa[arow * SA_STRIDE + acol + 8]) = ra1;
  };
#if USE_ASYNC_LDS
  auto asyncA = [&](int kk, int buf) {  // global -> LDS DMA, tracked by ASYNCcnt
    __builtin_amdgcn_global_load_async_to_lds_b128(
        (g_v4i*)(aptr + kk + acol),
        (l_v4i*)&sA[buf][arow * SA_STRIDE + acol], 0, 0);
    __builtin_amdgcn_global_load_async_to_lds_b128(
        (g_v4i*)(aptr + kk + acol + 8),
        (l_v4i*)&sA[buf][arow * SA_STRIDE + acol + 8], 0, 0);
  };
#endif
  auto fetchB = [&](int kk) {
    const float* g0 = WepBase + (long)(kk + bk2) * ND;
    rb0 = ((const float4*)g0)[0];
    rb1 = ((const float4*)g0)[1];
    rb2 = ((const float4*)(g0 + ND))[0];
    rb3 = ((const float4*)(g0 + ND))[1];
  };
  auto stageB = [&](int buf) {
    __bf16* sb = &sB[buf][0];
    float lo[8] = {rb0.x, rb0.y, rb0.z, rb0.w, rb1.x, rb1.y, rb1.z, rb1.w};
    float hi[8] = {rb2.x, rb2.y, rb2.z, rb2.w, rb3.x, rb3.y, rb3.z, rb3.w};
#pragma unroll
    for (int j = 0; j < 8; ++j) {
      unsigned pk = f2bfbits(lo[j]) | (f2bfbits(hi[j]) << 16);  // (k, k+1) pair
      *(unsigned*)(&sb[(bfc + j) * SA_STRIDE + bk2]) = pk;
    }
  };

  v8f zero8 = {0.f, 0.f, 0.f, 0.f, 0.f, 0.f, 0.f, 0.f};
  v8f acc[4][2];
#pragma unroll
  for (int mi = 0; mi < 4; ++mi)
#pragma unroll
    for (int ni = 0; ni < 2; ++ni) acc[mi][ni] = zero8;

  const int lrow  = lane & 15;
  const int koff  = (lane >> 4) * 8;   // A frag K interleave {0..7,16..23}/{8..15,24..31}
  const int kbase = (lane >> 4) * 8;   // B frag K interleave {0..7,16..23}/{8..15,24..31}

#if USE_ASYNC_LDS
  asyncA(0, 0);
  fetchB(0);
#else
  fetchA(0);
  fetchB(0);
#endif
  int buf = 0;
  for (int kk = 0; kk < KD; kk += 32) {
    const int kn = (kk + 32 < KD) ? kk + 32 : kk;
#if USE_ASYNC_LDS
    stageB(buf);
    __builtin_amdgcn_s_wait_asynccnt(0);  // own-wave A-tile DMA done (barrier only waits DScnt)
    __syncthreads();                      // all waves' tile-i A+B visible; buf^1 reads done
    asyncA(kn, buf ^ 1);                  // DMA next A tile while computing
    fetchB(kn);
#else
    stageA(buf);
    stageB(buf);
    __syncthreads();
    fetchA(kn);
    fetchB(kn);
#endif
    // pull the K-step after next toward L2/WGP$ (speculative, OOB-safe)
    __builtin_prefetch(aptr + kk + 64 + acol, 0, 3);
    __builtin_prefetch(WepBase + (long)(kk + 64 + bk2) * ND, 0, 3);

    v16bf af[4], bfr[2];
#pragma unroll
    for (int mi = 0; mi < 4; ++mi) {
      const __bf16* p = &sA[buf][(wm * 64 + mi * 16 + lrow) * SA_STRIDE];
      v8bf plo = *(const v8bf*)(p + koff);
      v8bf phi = *(const v8bf*)(p + 16 + koff);
      af[mi] = __builtin_shufflevector(plo, phi, 0, 1, 2, 3, 4, 5, 6, 7, 8, 9,
                                       10, 11, 12, 13, 14, 15);
    }
#pragma unroll
    for (int ni = 0; ni < 2; ++ni) {
      const __bf16* p = &sB[buf][(wn * 32 + ni * 16 + lrow) * SA_STRIDE + kbase];
      v8bf plo = *(const v8bf*)(p);
      v8bf phi = *(const v8bf*)(p + 16);
      bfr[ni] = __builtin_shufflevector(plo, phi, 0, 1, 2, 3, 4, 5, 6, 7, 8, 9,
                                        10, 11, 12, 13, 14, 15);
    }
#pragma unroll
    for (int mi = 0; mi < 4; ++mi)
#pragma unroll
      for (int ni = 0; ni < 2; ++ni)
        acc[mi][ni] = __builtin_amdgcn_wmma_f32_16x16x32_bf16(
            false, af[mi], false, bfr[ni], (short)0, acc[mi][ni], false, false);

    buf ^= 1;
  }

  // Epilogue: acc VGPR r, lanes 0-15 => M=r, N=lane; lanes 16-31 => M=r+8
  const int mh = (lane >> 4) * 8;
  float bv[2];
#pragma unroll
  for (int ni = 0; ni < 2; ++ni)
    bv[ni] = bias[e * ND + n0 + wn * 32 + ni * 16 + lrow];

#pragma unroll
  for (int mi = 0; mi < 4; ++mi) {
#pragma unroll
    for (int r = 0; r < 8; ++r) {
      const int gm = m0 + wm * 64 + mi * 16 + mh + r;
      if (gm >= gcnt) continue;
      const int slot = perm[gbase + gm];
#pragma unroll
      for (int ni = 0; ni < 2; ++ni) {
        const int ncol = n0 + wn * 32 + ni * 16 + lrow;
        float v = acc[mi][ni][r];
        if (STAGE_IN) {
          v += bv[ni];
          v = v > 0.f ? v : 0.f;
          Hout[(long)slot * ND + ncol] = f2bf(v);
        } else {
          const float p = topk_p[slot];
          Yout[(long)slot * ND + ncol] = p * (v + bv[ni]);
        }
      }
    }
  }
}

// ---------------- launch ----------------

extern "C" void kernel_launch(void* const* d_in, const int* in_sizes, int n_in,
                              void* d_out, int out_size, void* d_ws, size_t ws_size,
                              hipStream_t stream) {
  (void)in_sizes; (void)n_in; (void)out_size; (void)ws_size;
  const float* x     = (const float*)d_in[0];
  const float* Wr    = (const float*)d_in[1];
  const float* br    = (const float*)d_in[2];
  const float* W_in  = (const float*)d_in[3];
  const float* b_in  = (const float*)d_in[4];
  const float* W_out = (const float*)d_in[5];
  const float* b_out = (const float*)d_in[6];
  float* y = (float*)d_out;

  char* ws = (char*)d_ws;
  size_t off = 0;
  auto alloc = [&](size_t bytes) -> void* {
    void* p = ws + off;
    off = (off + bytes + 255) & ~(size_t)255;
    return p;
  };
  __bf16* xb     = (__bf16*)alloc((size_t)NTOK * DDIM * 2);   //  8 MB
  __bf16* habuf  = (__bf16*)alloc((size_t)NSLOT * FDIM * 2);  // 64 MB
  __bf16* hcomb  = (__bf16*)alloc((size_t)NTOK * FDIM * 2);   // 32 MB
  float*  yb     = (float*) alloc((size_t)NSLOT * DDIM * 4);  // 32 MB
  float*  tkp    = (float*) alloc((size_t)NSLOT * 4);
  int*    tke    = (int*)   alloc((size_t)NSLOT * 4);
  int*    perm   = (int*)   alloc((size_t)NSLOT * 4);
  int*    counts = (int*)   alloc(64);
  int*    offs   = (int*)   alloc(64);
  int*    cursor = (int*)   alloc(64);

  k_zero<<<1, 64, 0, stream>>>(counts);
  k_cvt_x<<<(NTOK * DDIM) / 1024, 256, 0, stream>>>(x, xb);
  k_router<<<NTOK / 8, 256, 0, stream>>>(x, Wr, br, tkp, tke, counts);
  k_scan<<<1, 32, 0, stream>>>(counts, offs, cursor);
  k_scatter<<<NSLOT / 256, 256, 0, stream>>>(tke, cursor, perm);

  moe_gemm<DDIM, FDIM, true><<<dim3(FDIM / 128, NTOK / 128, NEXP), 256, 0, stream>>>(
      xb, W_in, b_in, perm, offs, counts, tkp, habuf, nullptr);

  k_hcomb<<<NTOK, 256, 0, stream>>>(habuf, hcomb);

  moe_gemm<FDIM, DDIM, false><<<dim3(DDIM / 128, NTOK / 128, NEXP), 256, 0, stream>>>(
      hcomb, W_out, b_out, perm, offs, counts, tkp, nullptr, yb);

  k_ycomb<<<NTOK, 256, 0, stream>>>(yb, y);
}
